// ComponentGNN_75557064671345
// MI455X (gfx1250) — matrix operand order repaired and checked by reading, not compiled, not run on hardware
//
#include <hip/hip_runtime.h>

// ---------------------------------------------------------------------------
// Fused GAT GNN for MI455X (gfx1250, wave32, WMMA + async LDS copies).
// One workgroup per graph; whole graph resident in LDS.
// GEMMs via v_wmma_f32_16x16x32_f16; weights staged with
// global_load_async_to_lds_b128 (ASYNCcnt) from pre-transposed f16 images.
// ---------------------------------------------------------------------------

#define BGR      1024    // graphs
#define NODES    128     // nodes per graph
#define EPG      1024    // edges per graph (excl. self loops)
#define HCH      128     // hidden channels
#define HEADS    4
#define HD       32
#define LSTR     136     // f16 LDS row stride (68 dwords -> 4-bank shift/row)
#define NTHREADS 256     // 8 waves
#define WBYTES   (HCH * LSTR * 2)       // one weight image in bytes (34816)
#define WCHUNKS  (WBYTES / 16)          // 16B chunks per weight image (2176)

typedef _Float16 v16h __attribute__((ext_vector_type(16)));
typedef _Float16 v8h  __attribute__((ext_vector_type(8)));
typedef float    v8f  __attribute__((ext_vector_type(8)));

struct __align__(16) Smem {
    _Float16 hA[NODES * LSTR];   // activations (f16), input to GEMM
    _Float16 wB[HCH * LSTR];     // transposed weight tile Wt[n][k] (f16)
    _Float16 xh[NODES * LSTR];   // GEMM output (f16)
    float    ex[EPG * HEADS];    // per-edge exp(e - max) values
    float    exself[NODES * HEADS];
    float    ssrc[HEADS * NODES];
    float    sdst[HEADS * NODES];
    float    sinv[HEADS * NODES];
    float    hp[HCH];            // pooled features (sum over nodes)
    float    zbuf[64];           // MLP hidden
    int      csrOff[NODES + 1];
    int      csrSrc[EPG];
};

__device__ __forceinline__ float lrelu(float v) { return v > 0.f ? v : 0.2f * v; }

// LDS byte offset of a shared-memory pointer (generic -> addrspace(3) cast).
__device__ __forceinline__ unsigned lds_off(const void* p) {
    return (unsigned)(unsigned long long)(const __attribute__((address_space(3))) void*)p;
}

// CDNA5 async global->LDS copy, 16 bytes per lane. Tracked by ASYNCcnt.
__device__ __forceinline__ void async_copy_b128(unsigned lds_byte_off, const void* gaddr) {
    asm volatile("global_load_async_to_lds_b128 %0, %1, off"
                 :
                 : "v"(lds_byte_off), "v"((unsigned long long)gaddr)
                 : "memory");
}

__device__ __forceinline__ void wait_async0() {
    asm volatile("s_wait_asynccnt 0x0" ::: "memory");
}

// 128x128x128 GEMM: C = A * W, with Bt = W^T staged in LDS.
// Wave w computes rows [w*16, w*16+16), all 8 N-tiles, K-loop of 4 x 32.
__device__ __forceinline__ void gemm128_f16(const _Float16* A,
                                            const _Float16* Bt,
                                            _Float16* C,
                                            int wave, int lane) {
    const int l15   = lane & 15;
    const int hi    = lane >> 4;
    const int mbase = wave * 16;
    v8f acc[8] = {};
#pragma unroll
    for (int kb = 0; kb < HCH; kb += 32) {
        // A fragment (16x32 f16): lanes<16 hold K[kb..kb+7] & K[kb+16..kb+23],
        // lanes>=16 hold K[kb+8..kb+15] & K[kb+24..kb+31] (ISA 7.12.2).
        const _Float16* arow = A + (mbase + l15) * LSTR + kb + hi * 8;
        v8h alo = *(const v8h*)(arow);
        v8h ahi = *(const v8h*)(arow + 16);
        v16h a  = __builtin_shufflevector(alo, ahi, 0, 1, 2, 3, 4, 5, 6, 7,
                                          8, 9, 10, 11, 12, 13, 14, 15);
#pragma unroll
        for (int j = 0; j < 8; ++j) {
            // B fragment (32x16): N = lane&15, lanes<16 K[kb..kb+7] & K[kb+16..kb+23],
            // lanes>=16 K[kb+8..kb+15] & K[kb+24..kb+31] (same K order as A).
            const _Float16* brow = Bt + (j * 16 + l15) * LSTR + kb + hi * 8;
            v8h b0 = *(const v8h*)(brow);
            v8h b1 = *(const v8h*)(brow + 16);
            v16h b = __builtin_shufflevector(b0, b1, 0, 1, 2, 3, 4, 5, 6, 7,
                                             8, 9, 10, 11, 12, 13, 14, 15);
            acc[j] = __builtin_amdgcn_wmma_f32_16x16x32_f16(
                false, a, false, b, (short)0, acc[j], false, false);
        }
    }
    // C/D layout: element r of lane L -> row mbase + 8*(L>>4) + r, col j*16+(L&15)
#pragma unroll
    for (int j = 0; j < 8; ++j)
#pragma unroll
        for (int r = 0; r < 8; ++r)
            C[(mbase + hi * 8 + r) * LSTR + j * 16 + l15] = (_Float16)acc[j][r];
}

// ---------------------------------------------------------------------------
// Kernel 0: build CSR (dst-bucketed) from the shared per-graph edge list.
// Serial stable scatter -> deterministic.
// ---------------------------------------------------------------------------
__global__ void build_csr_kernel(const int* __restrict__ edge_index,
                                 int* __restrict__ csrOff,
                                 int* __restrict__ csrSrc) {
    __shared__ int cnt[NODES];
    __shared__ int off[NODES + 1];
    const int tid = threadIdx.x;
    if (tid < NODES) cnt[tid] = 0;
    __syncthreads();
    for (int e = tid; e < EPG; e += NTHREADS)
        atomicAdd(&cnt[edge_index[EPG + e]], 1);   // dst row
    __syncthreads();
    if (tid == 0) {
        int a = 0;
        for (int i = 0; i < NODES; ++i) { off[i] = a; a += cnt[i]; }
        off[NODES] = a;
        // stable serial scatter
        for (int i = 0; i < NODES; ++i) cnt[i] = off[i];
        for (int e = 0; e < EPG; ++e) {
            int d = edge_index[EPG + e];
            csrSrc[cnt[d]++] = edge_index[e];      // src row
        }
    }
    __syncthreads();
    if (tid <= NODES) csrOff[tid] = off[tid];
}

// ---------------------------------------------------------------------------
// Kernel 0b: pre-transpose + f16-convert both GAT weights into workspace,
// already laid out with the LDS row stride (padding zeroed) so the main
// kernel can async-memcpy them straight into LDS.
// ---------------------------------------------------------------------------
__global__ void prep_weights_kernel(const float* __restrict__ w0,
                                    const float* __restrict__ w1,
                                    _Float16* __restrict__ wt0,
                                    _Float16* __restrict__ wt1) {
    int t = blockIdx.x * blockDim.x + threadIdx.x;
    if (t >= HCH * LSTR) return;
    int n = t / LSTR, k = t % LSTR;
    _Float16 a = (_Float16)0.f, b = (_Float16)0.f;
    if (k < HCH) {
        a = (_Float16)w0[k * HCH + n];   // transpose: Wt[n][k] = W[k][n]
        b = (_Float16)w1[k * HCH + n];
    }
    wt0[t] = a;
    wt1[t] = b;
}

// ---------------------------------------------------------------------------
// Kernel 1: fused per-graph GNN.
// ---------------------------------------------------------------------------
__global__ __launch_bounds__(NTHREADS) void gnn_fused_kernel(
    const float* __restrict__ x,
    const int* __restrict__ csrOffG, const int* __restrict__ csrSrcG,
    const _Float16* __restrict__ wt0, const _Float16* __restrict__ wt1,
    const float* __restrict__ w_in, const float* __restrict__ b_in,
    const float* __restrict__ g0as, const float* __restrict__ g0ad,
    const float* __restrict__ g0b,
    const float* __restrict__ bn0g, const float* __restrict__ bn0b,
    const float* __restrict__ bn0m, const float* __restrict__ bn0v,
    const float* __restrict__ g1as, const float* __restrict__ g1ad,
    const float* __restrict__ g1b,
    const float* __restrict__ bn1g, const float* __restrict__ bn1b,
    const float* __restrict__ bn1m, const float* __restrict__ bn1v,
    const float* __restrict__ w1, const float* __restrict__ b1,
    const float* __restrict__ w2, const float* __restrict__ b2,
    float* __restrict__ out) {
    __shared__ Smem s;
    const int tid  = threadIdx.x;
    const int wave = tid >> 5;
    const int lane = tid & 31;
    const int g    = blockIdx.x;   // graph id

    // ---- Phase 1: issue async W0^T -> wB; CSR -> LDS; input proj -> hA ----
    {
        const unsigned wb = lds_off(s.wB);
        for (int i = tid; i < WCHUNKS; i += NTHREADS)
            async_copy_b128(wb + i * 16, wt0 + i * 8);
    }
    if (tid <= NODES) s.csrOff[tid] = csrOffG[tid];
    for (int e = tid; e < EPG; e += NTHREADS) s.csrSrc[e] = csrSrcG[e];
    for (int t = tid; t < NODES * HCH; t += NTHREADS) {
        int n = t >> 7, c = t & 127;
        float v = fmaf(x[g * NODES + n], w_in[c], b_in[c]);
        s.hA[n * LSTR + c] = (_Float16)fmaxf(v, 0.f);
    }
    wait_async0();
    __syncthreads();

    // ---- Phase 2: GEMM xh = hA @ W0 ----
    gemm128_f16(s.hA, s.wB, s.xh, wave, lane);
    __syncthreads();

    // ---- Phase 3: issue async W1^T -> wB (wB dead after GEMM1);
    //      attention scores (4 heads x 32) overlap the copy ----
    {
        const unsigned wb = lds_off(s.wB);
        for (int i = tid; i < WCHUNKS; i += NTHREADS)
            async_copy_b128(wb + i * 16, wt1 + i * 8);
    }
    for (int t = tid; t < HEADS * NODES; t += NTHREADS) {
        int h = t >> 7, n = t & 127;
        float ss = 0.f, sd = 0.f;
        for (int c = 0; c < HD; ++c) {
            float v = (float)s.xh[n * LSTR + h * HD + c];
            ss = fmaf(v, g0as[h * HD + c], ss);
            sd = fmaf(v, g0ad[h * HD + c], sd);
        }
        s.ssrc[t] = ss;
        s.sdst[t] = sd;
    }
    __syncthreads();

    // ---- Phase 4: per-dst softmax (incl. self loop) ----
    if (tid < NODES) {
        int n = tid, e0 = s.csrOff[n], e1 = s.csrOff[n + 1];
        float selfv[HEADS], mx[HEADS], sum[HEADS];
        for (int h = 0; h < HEADS; ++h) {
            selfv[h] = lrelu(s.ssrc[h * NODES + n] + s.sdst[h * NODES + n]);
            mx[h] = selfv[h];
        }
        for (int e = e0; e < e1; ++e) {
            int sn = s.csrSrc[e];
            for (int h = 0; h < HEADS; ++h) {
                float v = lrelu(s.ssrc[h * NODES + sn] + s.sdst[h * NODES + n]);
                s.ex[e * HEADS + h] = v;
                mx[h] = fmaxf(mx[h], v);
            }
        }
        for (int h = 0; h < HEADS; ++h) {
            float v = expf(selfv[h] - mx[h]);
            s.exself[n * HEADS + h] = v;
            sum[h] = v;
        }
        for (int e = e0; e < e1; ++e)
            for (int h = 0; h < HEADS; ++h) {
                float v = expf(s.ex[e * HEADS + h] - mx[h]);
                s.ex[e * HEADS + h] = v;
                sum[h] += v;
            }
        for (int h = 0; h < HEADS; ++h) s.sinv[h * NODES + n] = 1.f / sum[h];
    }
    __syncthreads();

    // ---- Phase 5: aggregate + bias + BN0 + ReLU -> hA (f16) ----
    for (int t = tid; t < NODES * HCH; t += NTHREADS) {
        int n = t >> 7, c = t & 127, h = c >> 5;
        int e0 = s.csrOff[n], e1 = s.csrOff[n + 1];
        float acc = s.exself[n * HEADS + h] * (float)s.xh[n * LSTR + c];
        for (int e = e0; e < e1; ++e)
            acc = fmaf(s.ex[e * HEADS + h], (float)s.xh[s.csrSrc[e] * LSTR + c], acc);
        float v = fmaf(acc, s.sinv[h * NODES + n], g0b[c]);
        v = fmaf((v - bn0m[c]) * rsqrtf(bn0v[c] + 1e-5f), bn0g[c], bn0b[c]);
        s.hA[n * LSTR + c] = (_Float16)fmaxf(v, 0.f);
    }
    wait_async0();          // W1^T copy finished (long overlapped)
    __syncthreads();

    // ---- Phase 6: GEMM xh = hA @ W1 ----
    gemm128_f16(s.hA, s.wB, s.xh, wave, lane);
    __syncthreads();

    // ---- Phase 7: single-head scores (128-d dot); zero hp ----
    for (int t = tid; t < 2 * NODES; t += NTHREADS) {
        int n = t & 127;
        const float* av = (t < NODES) ? g1as : g1ad;
        float acc = 0.f;
        for (int c = 0; c < HCH; ++c)
            acc = fmaf((float)s.xh[n * LSTR + c], av[c], acc);
        if (t < NODES) s.ssrc[n] = acc; else s.sdst[n] = acc;
    }
    if (tid < HCH) s.hp[tid] = 0.f;
    __syncthreads();

    // ---- Phase 8: per-dst softmax (1 head) ----
    if (tid < NODES) {
        int n = tid, e0 = s.csrOff[n], e1 = s.csrOff[n + 1];
        float selfv = lrelu(s.ssrc[n] + s.sdst[n]);
        float mx = selfv;
        for (int e = e0; e < e1; ++e) {
            float v = lrelu(s.ssrc[s.csrSrc[e]] + s.sdst[n]);
            s.ex[e] = v;
            mx = fmaxf(mx, v);
        }
        float v0 = expf(selfv - mx), sum = v0;
        s.exself[n] = v0;
        for (int e = e0; e < e1; ++e) {
            float v = expf(s.ex[e] - mx);
            s.ex[e] = v;
            sum += v;
        }
        s.sinv[n] = 1.f / sum;
    }
    __syncthreads();

    // ---- Phase 9: aggregate + bias + BN1 + ReLU + mean-pool (LDS atomics) ----
    for (int t = tid; t < NODES * HCH; t += NTHREADS) {
        int n = t >> 7, c = t & 127;
        int e0 = s.csrOff[n], e1 = s.csrOff[n + 1];
        float acc = s.exself[n] * (float)s.xh[n * LSTR + c];
        for (int e = e0; e < e1; ++e)
            acc = fmaf(s.ex[e], (float)s.xh[s.csrSrc[e] * LSTR + c], acc);
        float v = fmaf(acc, s.sinv[n], g1b[c]);
        v = fmaf((v - bn1m[c]) * rsqrtf(bn1v[c] + 1e-5f), bn1g[c], bn1b[c]);
        atomicAdd(&s.hp[c], fmaxf(v, 0.f));   // ds_add_f32
    }
    __syncthreads();

    // ---- Phase 10: MLP 128 -> 64 -> 1 ----
    if (tid < 64) {
        float a = b1[tid];
        for (int c = 0; c < HCH; ++c)
            a = fmaf(s.hp[c] * (1.f / 128.f), w1[c * 64 + tid], a);
        s.zbuf[tid] = fmaxf(a, 0.f);
    }
    __syncthreads();
    if (tid == 0) {
        float o = b2[0];
        for (int j = 0; j < 64; ++j) o = fmaf(s.zbuf[j], w2[j], o);
        out[g] = o;
    }
}

extern "C" void kernel_launch(void* const* d_in, const int* in_sizes, int n_in,
                              void* d_out, int out_size, void* d_ws, size_t ws_size,
                              hipStream_t stream) {
    (void)in_sizes; (void)n_in; (void)out_size; (void)ws_size;
    const float* x    = (const float*)d_in[0];
    const int*   ei   = (const int*)d_in[1];
    const float* w_in = (const float*)d_in[2];
    const float* b_in = (const float*)d_in[3];
    const float* g0w  = (const float*)d_in[4];
    const float* g0as = (const float*)d_in[5];
    const float* g0ad = (const float*)d_in[6];
    const float* g0b  = (const float*)d_in[7];
    const float* bn0g = (const float*)d_in[8];
    const float* bn0b = (const float*)d_in[9];
    const float* bn0m = (const float*)d_in[10];
    const float* bn0v = (const float*)d_in[11];
    const float* g1w  = (const float*)d_in[12];
    const float* g1as = (const float*)d_in[13];
    const float* g1ad = (const float*)d_in[14];
    const float* g1b  = (const float*)d_in[15];
    const float* bn1g = (const float*)d_in[16];
    const float* bn1b = (const float*)d_in[17];
    const float* bn1m = (const float*)d_in[18];
    const float* bn1v = (const float*)d_in[19];
    const float* w1   = (const float*)d_in[20];
    const float* b1   = (const float*)d_in[21];
    const float* w2   = (const float*)d_in[22];
    const float* b2   = (const float*)d_in[23];

    // Workspace layout
    int*      csrOff = (int*)d_ws;                               // 129 ints
    int*      csrSrc = csrOff + (NODES + 1);                     // 1024 ints
    _Float16* wt0    = (_Float16*)((char*)d_ws + 8192);          // 16B aligned
    _Float16* wt1    = wt0 + HCH * LSTR;

    build_csr_kernel<<<1, NTHREADS, 0, stream>>>(ei, csrOff, csrSrc);
    prep_weights_kernel<<<(HCH * LSTR + NTHREADS - 1) / NTHREADS, NTHREADS, 0, stream>>>(
        g0w, g1w, wt0, wt1);
    gnn_fused_kernel<<<BGR, NTHREADS, 0, stream>>>(
        x, csrOff, csrSrc, wt0, wt1, w_in, b_in,
        g0as, g0ad, g0b, bn0g, bn0b, bn0m, bn0v,
        g1as, g1ad, g1b, bn1g, bn1b, bn1m, bn1v,
        w1, b1, w2, b2, (float*)d_out);
}
